// MoELayer_69561290326691
// MI455X (gfx1250) — compile-verified, matrix operand order checked by reading, not run on hardware
//
#include <hip/hip_runtime.h>
#include <hip/hip_bf16.h>
#include <math.h>

// ---------------------------------------------------------------------------
// MoE (E=8, C=1024, I=4096, top-2) for gfx1250 / MI455X.
// Routed grouped-GEMM using v_wmma_f32_16x16x32_bf16 (wave32).
// ---------------------------------------------------------------------------

#define E_NUM   8
#define C_DIM   1024
#define I_DIM   4096
#define N_TOK   4096      // B*T = 2*2048
#define K_TOP   2
#define TOKTILE 16
#define ICHUNK  128       // I processed per LDS-staged chunk

typedef __attribute__((ext_vector_type(16))) __bf16 v16bf;
typedef __attribute__((ext_vector_type(8)))  float  v8f;

// d_out element offsets (f32 units), tuple return order:
// final [2,2048,1024] | full_expert_outputs [4096,8,1024] | logits [4096,8] | sel [4096,2]
#define OFF_FINAL  0
#define OFF_FEO    (4096u * 1024u)                 //  4,194,304
#define OFF_LOG    (OFF_FEO + 4096u * 8u * 1024u)  // 37,748,736
#define OFF_SEL    (OFF_LOG + 4096u * 8u)          // 37,781,504

// workspace layout (bytes)
#define WS_COUNTS  0                         // 8 ints
#define WS_ETOK    32                        // [E][N] ints = 128 KB
#define WS_RW      (32 + E_NUM * N_TOK * 4)  // [N][2] f32 routing weights

// Load 16 consecutive f32 and pack to a bf16 WMMA half-K fragment.
__device__ __forceinline__ v16bf load_bf16_frag(const float* __restrict__ p) {
    v16bf b;
#pragma unroll
    for (int j = 0; j < 4; ++j) {
        float4 f = ((const float4*)p)[j];
        b[4 * j + 0] = (__bf16)f.x;
        b[4 * j + 1] = (__bf16)f.y;
        b[4 * j + 2] = (__bf16)f.z;
        b[4 * j + 3] = (__bf16)f.w;
    }
    return b;
}

// ---------------------------------------------------------------------------
// 1) Router: logits, top-2, softmax, per-expert token compaction.
// ---------------------------------------------------------------------------
__global__ void moe_router_kernel(const float* __restrict__ x,
                                  const float* __restrict__ gw,
                                  float* __restrict__ logits_out,
                                  int* __restrict__ sel_out,
                                  float* __restrict__ rw_out,
                                  int* __restrict__ counts,
                                  int* __restrict__ etok) {
    __shared__ float gs[E_NUM][C_DIM];  // 32 KB
    const int tid = threadIdx.x;
    for (int i = tid; i < E_NUM * C_DIM; i += 256)
        gs[i >> 10][i & (C_DIM - 1)] = gw[i];
    __syncthreads();

    const int n = blockIdx.x * 256 + tid;
    const float* xr = x + (size_t)n * C_DIM;
    float acc[E_NUM];
#pragma unroll
    for (int e = 0; e < E_NUM; ++e) acc[e] = 0.f;

    for (int c = 0; c < C_DIM; c += 4) {
        float4 xv = *(const float4*)(xr + c);
#pragma unroll
        for (int e = 0; e < E_NUM; ++e)
            acc[e] += xv.x * gs[e][c] + xv.y * gs[e][c + 1] +
                      xv.z * gs[e][c + 2] + xv.w * gs[e][c + 3];
    }

#pragma unroll
    for (int e = 0; e < E_NUM; ++e) logits_out[(size_t)n * E_NUM + e] = acc[e];

    // top-2 (first occurrence on ties, matching jax.lax.top_k)
    int i0 = 0; float v0 = acc[0];
#pragma unroll
    for (int e = 1; e < E_NUM; ++e) if (acc[e] > v0) { v0 = acc[e]; i0 = e; }
    int i1 = -1; float v1 = -3.4e38f;
#pragma unroll
    for (int e = 0; e < E_NUM; ++e)
        if (e != i0 && acc[e] > v1) { v1 = acc[e]; i1 = e; }

    // softmax over [v0, v1] (v0 >= v1)
    float e1 = __expf(v1 - v0);
    float inv = 1.0f / (1.0f + e1);
    float w0 = inv, w1 = e1 * inv;

    sel_out[2 * n + 0] = i0;
    sel_out[2 * n + 1] = i1;
    rw_out[2 * n + 0] = w0;
    rw_out[2 * n + 1] = w1;

    int p0 = atomicAdd(&counts[i0], 1);
    etok[i0 * N_TOK + p0] = n;
    int p1 = atomicAdd(&counts[i1], 1);
    etok[i1 * N_TOK + p1] = n;
}

// ---------------------------------------------------------------------------
// 2) Expert FFN: per-expert token tiles, out = gelu(x*w1^T)*w2^T via WMMA.
//    grid = (256 tiles, 8 experts), block = 256 threads = 8 wave32s.
// ---------------------------------------------------------------------------
__global__ void __launch_bounds__(256)
moe_expert_kernel(const float* __restrict__ x,
                  const float* __restrict__ w1,
                  const float* __restrict__ w2,
                  const int* __restrict__ counts,
                  const int* __restrict__ etok,
                  float* __restrict__ feo) {
    const int e = blockIdx.y;
    const int cnt = counts[e];
    const int tile0 = blockIdx.x * TOKTILE;
    if (tile0 >= cnt) return;  // uniform early exit

    __shared__ __attribute__((aligned(32))) __bf16 xs[TOKTILE][C_DIM];  // 32 KB
    __shared__ __attribute__((aligned(32))) __bf16 hs[TOKTILE][ICHUNK]; //  4 KB
    __shared__ int toks[TOKTILE];

    const int tid  = threadIdx.x;
    const int wave = tid >> 5;          // 0..7
    const int lane = tid & 31;
    const int hi   = lane >> 4;         // K-half / row-group selector
    const int ln   = lane & 15;

    if (tid < TOKTILE) {
        int idx = tile0 + tid;
        toks[tid] = etok[e * N_TOK + min(idx, cnt - 1)];  // clamp; masked at store
    }
    __syncthreads();

    // Gather x tile -> LDS as bf16
    for (int i = tid; i < TOKTILE * C_DIM; i += 256) {
        int m = i >> 10, c = i & (C_DIM - 1);
        xs[m][c] = (__bf16)x[(size_t)toks[m] * C_DIM + c];
    }
    __syncthreads();

    v8f oacc[8];                        // 16 tokens x 128 channels per wave
#pragma unroll
    for (int t = 0; t < 8; ++t) oacc[t] = (v8f){};

    for (int ic = 0; ic < I_DIM; ic += ICHUNK) {
        // ---- Phase 1: h(16 x 128) = gelu(x_tile @ w1^T), wave owns 16 i-cols
        {
            v8f hacc = {};
            const int icol = ic + wave * 16 + ln;
            const float* w1col = w1 + ((size_t)e * I_DIM + icol) * C_DIM + hi * 16;
#pragma unroll 4
            for (int kb = 0; kb < C_DIM; kb += 32) {
                v16bf a = *(const v16bf*)(&xs[ln][kb + hi * 16]);
                v16bf b = load_bf16_frag(w1col + kb);
                hacc = __builtin_amdgcn_wmma_f32_16x16x32_bf16(
                    false, a, false, b, (short)0, hacc, false, false);
            }
#pragma unroll
            for (int r = 0; r < 8; ++r) {
                int m = 8 * hi + r;
                float v = hacc[r];
                float g = 0.5f * v * (1.0f + erff(v * 0.70710678f));  // exact gelu
                hs[m][wave * 16 + ln] = (__bf16)g;
            }
        }
        __syncthreads();

        // ---- Phase 2: out(16 x 128 per wave) += h @ w2^T over this I chunk
        {
            v16bf afr[4];
#pragma unroll
            for (int kb = 0; kb < 4; ++kb)
                afr[kb] = *(const v16bf*)(&hs[ln][kb * 32 + hi * 16]);
#pragma unroll
            for (int t = 0; t < 8; ++t) {
                const int ccol = wave * 128 + t * 16 + ln;
                const float* w2col =
                    w2 + ((size_t)e * C_DIM + ccol) * I_DIM + ic + hi * 16;
#pragma unroll
                for (int kb = 0; kb < 4; ++kb) {
                    v16bf b = load_bf16_frag(w2col + kb * 32);
                    oacc[t] = __builtin_amdgcn_wmma_f32_16x16x32_bf16(
                        false, afr[kb], false, b, (short)0, oacc[t], false, false);
                }
            }
        }
        __syncthreads();
    }

    // Scatter into full_expert_outputs[token, e, :]
    const int nvalid = cnt - tile0;
#pragma unroll
    for (int t = 0; t < 8; ++t) {
        const int ccol = wave * 128 + t * 16 + ln;
#pragma unroll
        for (int r = 0; r < 8; ++r) {
            int m = 8 * hi + r;
            if (m < nvalid)
                feo[((size_t)toks[m] * E_NUM + e) * C_DIM + ccol] = oacc[t][r];
        }
    }
}

// ---------------------------------------------------------------------------
// 3) Combine: final[n,:] = rw0*feo[n,e0,:] + rw1*feo[n,e1,:]
// ---------------------------------------------------------------------------
__global__ void moe_final_kernel(const float* __restrict__ feo,
                                 const int* __restrict__ sel,
                                 const float* __restrict__ rw,
                                 float* __restrict__ out) {
    const int n = blockIdx.x;
    const int e0 = sel[2 * n], e1 = sel[2 * n + 1];
    const float w0 = rw[2 * n], w1 = rw[2 * n + 1];
    const float* f0 = feo + ((size_t)n * E_NUM + e0) * C_DIM;
    const float* f1 = feo + ((size_t)n * E_NUM + e1) * C_DIM;
    float* o = out + (size_t)n * C_DIM;
    for (int c = threadIdx.x; c < C_DIM; c += blockDim.x)
        o[c] = w0 * f0[c] + w1 * f1[c];
}

// ---------------------------------------------------------------------------
extern "C" void kernel_launch(void* const* d_in, const int* in_sizes, int n_in,
                              void* d_out, int out_size, void* d_ws, size_t ws_size,
                              hipStream_t stream) {
    const float* x  = (const float*)d_in[0];  // [2,2048,1024]
    const float* gw = (const float*)d_in[1];  // [8,1024]
    const float* w1 = (const float*)d_in[2];  // [8,4096,1024]
    const float* w2 = (const float*)d_in[3];  // [8,1024,4096]

    float* out    = (float*)d_out;
    float* final_ = out + OFF_FINAL;
    float* feo    = out + OFF_FEO;
    float* logits = out + OFF_LOG;
    int*   sel    = (int*)(out + OFF_SEL);

    char* ws     = (char*)d_ws;
    int*   cnts  = (int*)(ws + WS_COUNTS);
    int*   etok  = (int*)(ws + WS_ETOK);
    float* rwgt  = (float*)(ws + WS_RW);

    // Re-zero per call (harness does not re-poison between replays).
    hipMemsetAsync(cnts, 0, E_NUM * sizeof(int), stream);
    hipMemsetAsync(feo, 0, (size_t)N_TOK * E_NUM * C_DIM * sizeof(float), stream);

    moe_router_kernel<<<N_TOK / 256, 256, 0, stream>>>(x, gw, logits, sel, rwgt,
                                                       cnts, etok);

    dim3 grid(N_TOK / TOKTILE, E_NUM);  // 256 x 8; most blocks early-exit
    moe_expert_kernel<<<grid, 256, 0, stream>>>(x, w1, w2, cnts, etok, feo);

    moe_final_kernel<<<N_TOK, 256, 0, stream>>>(feo, sel, rwgt, final_);
}
